// SimpleVectorQuantizer_13915694039130
// MI455X (gfx1250) — matrix operand order repaired and checked by reading, not compiled, not run on hardware
//
#include <hip/hip_runtime.h>
#include <hip/hip_bf16.h>

#define N_ROWS 32768      // B*T
#define DDIM   128
#define KCB    8192

#define WAVES  4
#define MT     4
#define ROWS_PER_WAVE (MT * 16)                  // 64
#define ROWS_PER_BLOCK (WAVES * ROWS_PER_WAVE)   // 256

// LDS tile: NT codewords x 128 bf16, padded to 136 ushorts (272B) per row
// so the 16B ds reads are bank-conflict free. Double buffered, hi+lo.
#define NT   32
#define LROW 136

typedef __attribute__((ext_vector_type(16))) __bf16 v16bf;
typedef __attribute__((ext_vector_type(8)))  float  v8f;
typedef __attribute__((ext_vector_type(4)))  int    vint4;

union BF16x16 {
    unsigned short s[16];
    uint4 u[2];
    v16bf v;
};

__device__ __forceinline__ unsigned short f2bf(float f) {
    unsigned u = __float_as_uint(f);
    u += 0x7FFFu + ((u >> 16) & 1u);          // round-to-nearest-even
    return (unsigned short)(u >> 16);
}
__device__ __forceinline__ float bf2f(unsigned short h) {
    return __uint_as_float(((unsigned)h) << 16);
}

// --- async global->LDS path (gfx1250), guarded so either toolchain compiles
#if defined(__has_builtin)
#if __has_builtin(__builtin_amdgcn_global_load_async_to_lds_b128)
#define HAVE_ASYNC_LDS 1
#endif
#endif

#ifdef HAVE_ASYNC_LDS
typedef __attribute__((address_space(1))) vint4 as1_vint4;   // global int4
typedef __attribute__((address_space(3))) vint4 as3_vint4;   // LDS int4
__device__ __forceinline__ void async_copy16(const void* g, void* l) {
    __builtin_amdgcn_global_load_async_to_lds_b128(
        (as1_vint4*)(unsigned long long)g,
        (as3_vint4*)(unsigned)(unsigned long long)l, 0, 0);
}
__device__ __forceinline__ void async_wait0() {
#if __has_builtin(__builtin_amdgcn_s_wait_asynccnt)
    __builtin_amdgcn_s_wait_asynccnt(0);
#else
    asm volatile("s_wait_asynccnt 0x0" ::: "memory");
#endif
}
#else
__device__ __forceinline__ void async_copy16(const void* g, void* l) {
    *(uint4*)l = *(const uint4*)g;            // sync fallback: load + ds_store
}
__device__ __forceinline__ void async_wait0() {}
#endif

// ---------------------------------------------------------------------------
// Kernel 1: split codebooks into bf16 hi/lo, compute ||c||^2 per codeword.
// ---------------------------------------------------------------------------
__global__ void vq_prep_codebook(const float* __restrict__ cb,
                                 unsigned short* __restrict__ ch,
                                 unsigned short* __restrict__ cl,
                                 float* __restrict__ cnorm) {
    const int k = blockIdx.x;
    const int d = threadIdx.x;
    float c = cb[(size_t)k * DDIM + d];
    unsigned short h = f2bf(c);
    ch[(size_t)k * DDIM + d] = h;
    cl[(size_t)k * DDIM + d] = f2bf(c - bf2f(h));

    __shared__ float sred[DDIM];
    sred[d] = c * c;
    __syncthreads();
    for (int off = DDIM / 2; off > 0; off >>= 1) {
        if (d < off) sred[d] += sred[d + off];
        __syncthreads();
    }
    if (d == 0) cnorm[k] = sred[0];
}

// ---------------------------------------------------------------------------
// Kernel 2: WMMA GEMM (score = ||c||^2 - 2 x.c) + running argmin per row.
// Split-precision bf16: dot ~= xh*ch + xh*cl + xl*ch  (near-f32 accuracy).
// B tiles staged global->LDS (async, double-buffered, 32 codewords/tile),
// shared by all 4 waves; each wave consumes two 16-column sub-tiles.
// ---------------------------------------------------------------------------
__global__ void __launch_bounds__(WAVES * 32)
vq_argmin(const float* __restrict__ x,
          const unsigned short* __restrict__ ch,
          const unsigned short* __restrict__ cl,
          const float* __restrict__ cnorm,
          const unsigned char* __restrict__ mask,
          int* __restrict__ ind_out) {
    const int tid  = threadIdx.x;
    const int lane = tid & 31;
    const int lrow = lane & 15;
    const int half = lane >> 4;                    // 0 or 1
    const int rowbase = blockIdx.x * ROWS_PER_BLOCK + (tid >> 5) * ROWS_PER_WAVE;

    __shared__ __align__(16) unsigned short sHi[2][NT][LROW];
    __shared__ __align__(16) unsigned short sLo[2][NT][LROW];

    // ---- Resident A fragments (xh / xl): MT row-tiles x 4 k-blocks.
    // ISA A layout (16-bit 16x32): lane<16 -> K = {c0..c0+7, c0+16..c0+23}, c0=0;
    // lane>=16 -> c0=8.  Elements 2j/2j+1 pack into VGPR j.
    v16bf Ah[MT][4], Al[MT][4];
    #pragma unroll
    for (int mt = 0; mt < MT; ++mt) {
        const float* xr = x + (size_t)(rowbase + mt * 16 + lrow) * DDIM;
        #pragma unroll
        for (int kb = 0; kb < 4; ++kb) {
            const int c0 = kb * 32 + half * 8;
            float4 f0 = *(const float4*)(xr + c0);
            float4 f1 = *(const float4*)(xr + c0 + 4);
            float4 f2 = *(const float4*)(xr + c0 + 16);
            float4 f3 = *(const float4*)(xr + c0 + 20);
            float vals[16] = {f0.x, f0.y, f0.z, f0.w, f1.x, f1.y, f1.z, f1.w,
                              f2.x, f2.y, f2.z, f2.w, f3.x, f3.y, f3.z, f3.w};
            BF16x16 ah, al;
            #pragma unroll
            for (int j = 0; j < 16; ++j) {
                unsigned short h = f2bf(vals[j]);
                ah.s[j] = h;
                al.s[j] = f2bf(vals[j] - bf2f(h));
            }
            Ah[mt][kb] = ah.v;
            Al[mt][kb] = al.v;
        }
    }

    float best[MT][8];
    int   bidx[MT][8];
    #pragma unroll
    for (int mt = 0; mt < MT; ++mt)
        #pragma unroll
        for (int i = 0; i < 8; ++i) { best[mt][i] = 3.4e38f; bidx[mt][i] = 0; }

    // Stage helper: 128 threads move NT=32 codewords (hi+lo) = 16KB per tile.
    // Chunk c (0..511): col = c>>4, 16-byte segment seg = c&15.
    auto stage = [&](int buf, int nb) {
        #pragma unroll
        for (int r = 0; r < 4; ++r) {
            int c   = tid + r * 128;
            int col = c >> 4;
            int seg = c & 15;
            const unsigned short* gH = ch + (size_t)(nb + col) * DDIM + seg * 8;
            const unsigned short* gL = cl + (size_t)(nb + col) * DDIM + seg * 8;
            async_copy16(gH, &sHi[buf][col][seg * 8]);
            async_copy16(gL, &sLo[buf][col][seg * 8]);
        }
    };

    const int d0 = half * 8;    // B layout: lane<16 -> K={0..7,16..23}, lane>=16 -> K={8..15,24..31}

    stage(0, 0);
    int buf = 0;
    for (int nb = 0; nb < KCB; nb += NT) {
        async_wait0();
        __syncthreads();                       // staged tile visible to all waves
        if (nb + NT < KCB) stage(buf ^ 1, nb + NT);

        #pragma unroll
        for (int st = 0; st < NT / 16; ++st) {
            const int col = nb + st * 16 + lrow;
            const unsigned short* bHrow = &sHi[buf][st * 16 + lrow][0];
            const unsigned short* bLrow = &sLo[buf][st * 16 + lrow][0];
            v16bf Bh[4], Bl[4];
            #pragma unroll
            for (int kb = 0; kb < 4; ++kb) {
                BF16x16 t;
                t.u[0] = *(const uint4*)(bHrow + kb * 32 + d0);
                t.u[1] = *(const uint4*)(bHrow + kb * 32 + d0 + 16);
                Bh[kb] = t.v;
                t.u[0] = *(const uint4*)(bLrow + kb * 32 + d0);
                t.u[1] = *(const uint4*)(bLrow + kb * 32 + d0 + 16);
                Bl[kb] = t.v;
            }
            const float cn = cnorm[col];

            #pragma unroll
            for (int mt = 0; mt < MT; ++mt) {
                v8f acc = {0.f, 0.f, 0.f, 0.f, 0.f, 0.f, 0.f, 0.f};
                #pragma unroll
                for (int kb = 0; kb < 4; ++kb) {
                    acc = __builtin_amdgcn_wmma_f32_16x16x32_bf16(
                            false, Ah[mt][kb], false, Bh[kb], (short)0, acc, false, false);
                    acc = __builtin_amdgcn_wmma_f32_16x16x32_bf16(
                            false, Ah[mt][kb], false, Bl[kb], (short)0, acc, false, false);
                    acc = __builtin_amdgcn_wmma_f32_16x16x32_bf16(
                            false, Al[mt][kb], false, Bh[kb], (short)0, acc, false, false);
                }
                #pragma unroll
                for (int i = 0; i < 8; ++i) {
                    float s = fmaf(-2.0f, acc[i], cn);
                    bool lt = s < best[mt][i];
                    best[mt][i] = lt ? s   : best[mt][i];
                    bidx[mt][i] = lt ? col : bidx[mt][i];
                }
            }
        }
        __syncthreads();                       // all waves done reading buf
        buf ^= 1;
    }

    // ---- Cross-lane argmin within each 16-lane half (ties -> smallest index)
    #pragma unroll
    for (int mt = 0; mt < MT; ++mt) {
        #pragma unroll
        for (int i = 0; i < 8; ++i) {
            float b = best[mt][i];
            int  ix = bidx[mt][i];
            #pragma unroll
            for (int dd = 1; dd < 16; dd <<= 1) {
                float ob = __shfl_xor(b, dd, 32);
                int   oi = __shfl_xor(ix, dd, 32);
                if (ob < b || (ob == b && oi < ix)) { b = ob; ix = oi; }
            }
            if (lrow == 0) {
                int row = rowbase + mt * 16 + half * 8 + i;  // C layout: lanes>=16 hold M=i+8
                ind_out[row] = mask[row] ? ix : 0;
            }
        }
    }
}

// ---------------------------------------------------------------------------
// Kernel 3: gather codeword, write quantized*m, per-row loss contribution.
// commit(0.25) + codebook(1.0) losses are numerically 1.25*mean((q-x)^2)*m.
// ---------------------------------------------------------------------------
__global__ void vq_finalize(const float* __restrict__ x,
                            const float* __restrict__ cb,
                            const unsigned char* __restrict__ mask,
                            const int* __restrict__ ind,
                            float* __restrict__ qout,
                            float* __restrict__ per_loss) {
    const int row = blockIdx.x;
    const int d = threadIdx.x;
    const int idx = ind[row];
    const float m = mask[row] ? 1.0f : 0.0f;
    float q  = cb[(size_t)idx * DDIM + d];
    float xv = x[(size_t)row * DDIM + d];
    qout[(size_t)row * DDIM + d] = q * m;
    float diff = q - xv;

    __shared__ float sred[DDIM];
    sred[d] = diff * diff;
    __syncthreads();
    for (int off = DDIM / 2; off > 0; off >>= 1) {
        if (d < off) sred[d] += sred[d + off];
        __syncthreads();
    }
    if (d == 0) per_loss[row] = 1.25f * (sred[0] * (1.0f / DDIM)) * m;
}

// ---------------------------------------------------------------------------
// Kernel 4: deterministic masked-mean reduction of the loss.
// ---------------------------------------------------------------------------
__global__ void vq_loss_reduce(const float* __restrict__ per_loss,
                               const unsigned char* __restrict__ mask,
                               float* __restrict__ loss_out) {
    __shared__ float ss[256], sm[256];
    const int tid = threadIdx.x;
    float s = 0.0f, msum = 0.0f;
    for (int i = tid; i < N_ROWS; i += 256) {
        s += per_loss[i];
        msum += mask[i] ? 1.0f : 0.0f;
    }
    ss[tid] = s; sm[tid] = msum;
    __syncthreads();
    for (int off = 128; off > 0; off >>= 1) {
        if (tid < off) { ss[tid] += ss[tid + off]; sm[tid] += sm[tid + off]; }
        __syncthreads();
    }
    if (tid == 0) loss_out[0] = ss[0] / fmaxf(sm[0], 1.0f);
}

// ---------------------------------------------------------------------------
extern "C" void kernel_launch(void* const* d_in, const int* in_sizes, int n_in,
                              void* d_out, int out_size, void* d_ws, size_t ws_size,
                              hipStream_t stream) {
    const float*         x    = (const float*)d_in[0];
    const unsigned char* mask = (const unsigned char*)d_in[1];   // jnp bool = 1 byte
    const float*         cb   = (const float*)d_in[2];

    char* ws = (char*)d_ws;
    unsigned short* ch    = (unsigned short*)ws;                                  // 2 MB
    unsigned short* cl    = (unsigned short*)(ws + (size_t)KCB * DDIM * 2);       // 2 MB
    float*          cnorm = (float*)(ws + (size_t)KCB * DDIM * 4);                // 32 KB
    float*          ploss = (float*)(ws + (size_t)KCB * DDIM * 4 + KCB * 4);      // 128 KB

    // Output layout: ind (32768 int32) | quantized (32768*128 f32) | loss (1 f32)
    int*   ind_out  = (int*)d_out;
    float* qout     = (float*)d_out + N_ROWS;
    float* loss_out = (float*)d_out + N_ROWS + (size_t)N_ROWS * DDIM;

    vq_prep_codebook<<<KCB, DDIM, 0, stream>>>(cb, ch, cl, cnorm);
    vq_argmin<<<N_ROWS / ROWS_PER_BLOCK, WAVES * 32, 0, stream>>>(
        x, ch, cl, cnorm, mask, ind_out);
    vq_finalize<<<N_ROWS, DDIM, 0, stream>>>(x, cb, mask, ind_out, qout, ploss);
    vq_loss_reduce<<<1, 256, 0, stream>>>(ploss, mask, loss_out);
}
